// AttentionBlock_89154931130623
// MI455X (gfx1250) — compile-verified, matrix operand order checked by reading, not run on hardware
//
#include <hip/hip_runtime.h>
#include <hip/hip_bf16.h>

// ---------------------------------------------------------------------------
// AttentionBlock (Restormer-style MDTA cross-attention) for MI455X / gfx1250.
// All heavy GEMMs routed through v_wmma_f32_16x16x32_f16 (wave32).
// ---------------------------------------------------------------------------

typedef __attribute__((ext_vector_type(16))) _Float16 v16h;
typedef __attribute__((ext_vector_type(8)))  _Float16 v8h;
typedef __attribute__((ext_vector_type(8)))  float    v8f;

static constexpr int Bq   = 4;
static constexpr int Cq   = 256;
static constexpr int Hq   = 128;
static constexpr int Wq   = 128;
static constexpr int Nq   = Hq * Wq;      // 16384
static constexpr int HEADS = 8;
static constexpr int HD    = Cq / HEADS;  // 32

// ---- WMMA fragment helpers -------------------------------------------------
// 16-bit A/B fragment layout (ISA 7.12.2): lane half selects K-base (0 or 8),
// elements 0..7 -> K = kb+e, elements 8..15 -> K = kb+16+e. Row/col = lane%16.
__device__ __forceinline__ v16h frag_load(const _Float16* p) {
    int kb = (threadIdx.x & 16) ? 8 : 0;
    union { v16h v; v8h h[2]; } u;
    u.h[0] = *(const v8h*)(p + kb);
    u.h[1] = *(const v8h*)(p + kb + 16);
    return u.v;
}

__device__ __forceinline__ v8f wmma_f16(v16h a, v16h b, v8f c) {
    return __builtin_amdgcn_wmma_f32_16x16x32_f16(false, a, false, b,
                                                  (short)0, c, false, false);
}

// ---- weight repack to f16 --------------------------------------------------
__global__ void cvt_w_kernel(const float* __restrict__ wkv,
                             const float* __restrict__ wq,
                             const float* __restrict__ wp,
                             _Float16* __restrict__ wkv_h,
                             _Float16* __restrict__ wq_h,
                             _Float16* __restrict__ wp_h) {
    int i = blockIdx.x * 256 + threadIdx.x;
    if (i < 2 * Cq * Cq) wkv_h[i] = (_Float16)wkv[i];
    if (i < Cq * Cq)     wp_h[i]  = (_Float16)wp[i];
    if (i < Cq * Cq * 9) {                 // OIHW -> [tap][o][c]
        int oc = i / 9, t = i % 9;
        wq_h[t * (Cq * Cq) + oc] = (_Float16)wq[i];
    }
}

// ---- LayerNorm over channels; emits channels-last f16 (+optional f32 NCHW) -
__global__ void ln_kernel(const float* __restrict__ in,
                          const float* __restrict__ w,
                          const float* __restrict__ bias,
                          _Float16* __restrict__ out_cl,   // [B,N,C] f16
                          float* __restrict__ out_f32) {   // [B,C,N] or null
    int pid = blockIdx.x * 256 + threadIdx.x;   // b*N + n
    int b = pid >> 14, n = pid & (Nq - 1);
    const float* src = in + (size_t)b * Cq * Nq + n;
    float s1 = 0.f, s2 = 0.f;
    for (int c = 0; c < Cq; ++c) {
        float v = src[(size_t)c * Nq];
        s1 += v; s2 += v * v;
    }
    float mu  = s1 * (1.f / Cq);
    float var = s2 * (1.f / Cq) - mu * mu;
    float rin = rsqrtf(var + 1e-5f);
    _Float16* drow = out_cl + (size_t)pid * Cq;
    for (int c = 0; c < Cq; ++c) {
        float v = (src[(size_t)c * Nq] - mu) * rin * w[c] + bias[c];
        drow[c] = (_Float16)v;
        if (out_f32) out_f32[(size_t)b * Cq * Nq + (size_t)c * Nq + n] = v;
    }
}

// ---- generic 1x1-conv GEMM: out[b,o,n] = sum_c A[o,c] * Bcl[b,n,c] ---------
// one wave computes a 16(M) x 64(N) tile; K = 256.
__global__ void gemm1x1_kernel(const _Float16* __restrict__ A,   // [M,256]
                               const _Float16* __restrict__ Bcl, // [B,N,256]
                               float* __restrict__ outF,         // [B,M,N] or null
                               _Float16* __restrict__ outH,      // [B,M,N] or null
                               const float* __restrict__ res,    // [B,256,N] or null
                               int M) {
    int wid  = (blockIdx.x * blockDim.x + threadIdx.x) >> 5;
    int lane = threadIdx.x & 31, r15 = lane & 15;
    int n64  = Nq / 64;
    int tiles = (M >> 4) * n64;
    int b  = wid / tiles;
    int t  = wid % tiles;
    int o0 = (t / n64) << 4;
    int n0 = (t % n64) << 6;

    v8f acc[4] = {};
    const _Float16* Arow  = A + (size_t)(o0 + r15) * Cq;
    const _Float16* Bbase = Bcl + (size_t)b * Nq * Cq;
    for (int k0 = 0; k0 < Cq; k0 += 32) {
        __builtin_prefetch(Arow + k0 + 32, 0, 1);
        v16h a = frag_load(Arow + k0);
#pragma unroll
        for (int j = 0; j < 4; ++j) {
            int n = n0 + j * 16 + r15;
            v16h bb = frag_load(Bbase + (size_t)n * Cq + k0);
            acc[j] = wmma_f16(a, bb, acc[j]);
        }
    }
    int coff = (lane & 16) ? 8 : 0;
#pragma unroll
    for (int j = 0; j < 4; ++j)
        for (int r = 0; r < 8; ++r) {
            int o = o0 + coff + r;
            int n = n0 + j * 16 + r15;
            float v = acc[j][r];
            if (res)  v += res[((size_t)b * Cq + o) * Nq + n];
            if (outF) outF[((size_t)b * M + o) * Nq + n] = v;
            if (outH) outH[((size_t)b * M + o) * Nq + n] = (_Float16)v;
        }
}

// ---- depthwise 3x3 (SAME) on kv; writes K (ch-major) and V ([b,h,n,d]) -----
__global__ void dw_kernel(const _Float16* __restrict__ kv,   // [B,512,N]
                          const float* __restrict__ wdw,     // [512,1,3,3]
                          _Float16* __restrict__ kout,       // [B,256,N]
                          _Float16* __restrict__ vout) {     // [B,8,N,32]
    size_t idx = (size_t)blockIdx.x * 256 + threadIdx.x;     // B*512*N
    int n  = idx & (Nq - 1);
    size_t bc = idx >> 14;
    int ch = bc & 511, b = (int)(bc >> 9);
    int x = n & (Wq - 1), y = n >> 7;
    const float* wp = wdw + ch * 9;
    const _Float16* src = kv + (bc << 14);
    float acc = 0.f;
#pragma unroll
    for (int dy = -1; dy <= 1; ++dy) {
        int yy = y + dy;
        if ((unsigned)yy >= (unsigned)Hq) continue;
#pragma unroll
        for (int dx = -1; dx <= 1; ++dx) {
            int xx = x + dx;
            if ((unsigned)xx >= (unsigned)Wq) continue;
            acc += wp[(dy + 1) * 3 + (dx + 1)] * (float)src[yy * Wq + xx];
        }
    }
    if (ch < Cq) {
        kout[((size_t)b * Cq + ch) * Nq + n] = (_Float16)acc;
    } else {
        int cc = ch - Cq, h = cc >> 5, d = cc & 31;
        vout[(((size_t)(b * HEADS + h)) * Nq + n) * HD + d] = (_Float16)acc;
    }
}

// ---- q 3x3 conv as 9-tap WMMA GEMM accumulation ----------------------------
__global__ void qconv_kernel(const _Float16* __restrict__ Wq9,  // [9,256,256]
                             const _Float16* __restrict__ yn_cl,// [B,N,256]
                             _Float16* __restrict__ qout) {     // [B,256,N]
    int wid  = (blockIdx.x * blockDim.x + threadIdx.x) >> 5;
    int lane = threadIdx.x & 31, r15 = lane & 15;
    int b  = wid >> 12;            // 16 * 256 tiles per batch
    int t  = wid & 4095;
    int o0 = (t >> 8) << 4;
    int n0 = (t & 255) << 6;
    int y  = n0 >> 7, xb = n0 & (Wq - 1);

    v8f acc[4] = {};
    const _Float16* ybase = yn_cl + (size_t)b * Nq * Cq;
    for (int tap = 0; tap < 9; ++tap) {
        int dy = tap / 3 - 1, dx = tap % 3 - 1;
        int yy = y + dy;
        if ((unsigned)yy >= (unsigned)Hq) continue;   // SAME zero-pad row
        int nshift = dy * Wq + dx;
        const _Float16* Arow = Wq9 + (size_t)tap * (Cq * Cq) + (size_t)(o0 + r15) * Cq;
        for (int k0 = 0; k0 < Cq; k0 += 32) {
            v16h a = frag_load(Arow + k0);
#pragma unroll
            for (int j = 0; j < 4; ++j) {
                int xx = xb + j * 16 + r15 + dx;
                v16h bb = {};
                if ((unsigned)xx < (unsigned)Wq) {
                    int n = n0 + j * 16 + r15 + nshift;
                    bb = frag_load(ybase + (size_t)n * Cq + k0);
                }
                acc[j] = wmma_f16(a, bb, acc[j]);
            }
        }
    }
    int coff = (lane & 16) ? 8 : 0;
#pragma unroll
    for (int j = 0; j < 4; ++j)
        for (int r = 0; r < 8; ++r) {
            int o = o0 + coff + r;
            int n = n0 + j * 16 + r15;
            qout[((size_t)b * Cq + o) * Nq + n] = (_Float16)acc[j][r];
        }
}

// ---- per-row sum of squares (for q/k L2 norms over N) ----------------------
__global__ void rowsumsq_kernel(const _Float16* __restrict__ m,
                                float* __restrict__ out, int rl) {
    __shared__ float red[256];
    size_t row = blockIdx.x;
    float s = 0.f;
    for (int i = threadIdx.x; i < rl; i += 256) {
        float v = (float)m[row * rl + i];
        s += v * v;
    }
    red[threadIdx.x] = s;
    __syncthreads();
    for (int st = 128; st > 0; st >>= 1) {
        if ((int)threadIdx.x < st) red[threadIdx.x] += red[threadIdx.x + st];
        __syncthreads();
    }
    if (threadIdx.x == 0) out[row] = red[0];
}

// ---- attention scores + softmax: one block (8 waves) per (b,head) ----------
__global__ void attn_kernel(const _Float16* __restrict__ q,  // [B,256,N]
                            const _Float16* __restrict__ k,  // [B,256,N]
                            const float* __restrict__ qn2,   // [B*256]
                            const float* __restrict__ kn2,
                            const float* __restrict__ temp,  // [8]
                            _Float16* __restrict__ P) {      // [B*8,32,32]
    __shared__ float sS[8][HD * HD];
    int bh = blockIdx.x;
    int b = bh >> 3, h = bh & 7;
    int lane = threadIdx.x & 31, w = threadIdx.x >> 5, r15 = lane & 15;
    const _Float16* qb = q + ((size_t)b * Cq + h * HD) * Nq;
    const _Float16* kb = k + ((size_t)b * Cq + h * HD) * Nq;

    v8f acc[4] = {};
    for (int k0 = w * 32; k0 < Nq; k0 += 8 * 32) {
        v16h alo = frag_load(qb + (size_t)r15 * Nq + k0);
        v16h ahi = frag_load(qb + (size_t)(r15 + 16) * Nq + k0);
        v16h blo = frag_load(kb + (size_t)r15 * Nq + k0);
        v16h bhi = frag_load(kb + (size_t)(r15 + 16) * Nq + k0);
        acc[0] = wmma_f16(alo, blo, acc[0]);
        acc[1] = wmma_f16(alo, bhi, acc[1]);
        acc[2] = wmma_f16(ahi, blo, acc[2]);
        acc[3] = wmma_f16(ahi, bhi, acc[3]);
    }
    int coff = (lane & 16) ? 8 : 0;
#pragma unroll
    for (int f = 0; f < 4; ++f)
        for (int r = 0; r < 8; ++r) {
            int c = ((f >> 1) * 16) + coff + r;
            int d = ((f & 1) * 16) + r15;
            sS[w][c * HD + d] = acc[f][r];
        }
    __syncthreads();

    for (int i = threadIdx.x; i < HD * HD; i += 256) {
        float s = 0.f;
        for (int ww = 0; ww < 8; ++ww) s += sS[ww][i];
        int c = i >> 5, d = i & 31;
        float nqv = fmaxf(sqrtf(qn2[b * Cq + h * HD + c]), 1e-12f);
        float nkv = fmaxf(sqrtf(kn2[b * Cq + h * HD + d]), 1e-12f);
        sS[0][i] = s / (nqv * nkv) * temp[h];
    }
    __syncthreads();

    if (threadIdx.x < HD) {
        int c = threadIdx.x;
        float mx = -1e30f;
        for (int d = 0; d < HD; ++d) mx = fmaxf(mx, sS[0][c * HD + d]);
        float sum = 0.f;
        for (int d = 0; d < HD; ++d) sum += __expf(sS[0][c * HD + d] - mx);
        float inv = 1.f / sum;
        for (int d = 0; d < HD; ++d)
            P[((size_t)bh * HD + c) * HD + d] =
                (_Float16)(__expf(sS[0][c * HD + d] - mx) * inv);
    }
}

// ---- out = P @ V, written channels-last f16 for the proj GEMM --------------
__global__ void av_kernel(const _Float16* __restrict__ P,   // [B*8,32,32]
                          const _Float16* __restrict__ Vt,  // [B,8,N,32]
                          _Float16* __restrict__ attn_cl) { // [B,N,256]
    int wid  = (blockIdx.x * 256 + threadIdx.x) >> 5;
    int lane = threadIdx.x & 31, r15 = lane & 15;
    int bh = wid >> 8;
    int n0 = (wid & 255) << 6;
    int b = bh >> 3, h = bh & 7;
    const _Float16* Pb = P + (size_t)bh * (HD * HD);
    v16h alo = frag_load(Pb + (size_t)r15 * HD);
    v16h ahi = frag_load(Pb + (size_t)(r15 + 16) * HD);
    int coff = (lane & 16) ? 8 : 0;
#pragma unroll
    for (int j = 0; j < 4; ++j) {
        int n = n0 + j * 16 + r15;
        v16h bb = frag_load(Vt + ((size_t)bh * Nq + n) * HD);
        v8f zlo = {}, zhi = {};
        v8f lo = wmma_f16(alo, bb, zlo);
        v8f hi = wmma_f16(ahi, bb, zhi);
        v8h p0, p1;
        for (int r = 0; r < 8; ++r) {
            p0[r] = (_Float16)lo[r];
            p1[r] = (_Float16)hi[r];
        }
        _Float16* dst = attn_cl + ((size_t)b * Nq + n) * Cq + h * HD;
        *(v8h*)(dst + coff)      = p0;   // channels c0..c0+7  (rows 0-15 tile)
        *(v8h*)(dst + 16 + coff) = p1;   // channels 16+...    (rows 16-31 tile)
    }
}

// ---------------------------------------------------------------------------
extern "C" void kernel_launch(void* const* d_in, const int* in_sizes, int n_in,
                              void* d_out, int out_size, void* d_ws, size_t ws_size,
                              hipStream_t stream) {
    const float* x       = (const float*)d_in[0];
    const float* y       = (const float*)d_in[1];
    const float* ln_kv_w = (const float*)d_in[2];
    const float* ln_kv_b = (const float*)d_in[3];
    const float* ln_q_w  = (const float*)d_in[4];
    const float* ln_q_b  = (const float*)d_in[5];
    const float* W_kv    = (const float*)d_in[6];
    const float* W_dw    = (const float*)d_in[7];
    const float* W_q     = (const float*)d_in[8];
    const float* W_proj  = (const float*)d_in[9];
    const float* temp    = (const float*)d_in[10];
    float* out = (float*)d_out;

    char* ws = (char*)d_ws;
    size_t off = 0;
    auto alloc = [&](size_t bytes) -> void* {
        void* p = ws + off;
        off = (off + bytes + 255) & ~(size_t)255;
        return p;
    };
    const size_t BNC = (size_t)Bq * Nq * Cq;           // 16.8M elements
    _Float16* xn_h   = (_Float16*)alloc(BNC * 2);      // [B,N,C]
    _Float16* yn_h   = (_Float16*)alloc(BNC * 2);      // [B,N,C]
    float*    yn_f   = (float*)   alloc(BNC * 4);      // [B,C,N]
    _Float16* kv_h   = (_Float16*)alloc(BNC * 4);      // [B,512,N]
    _Float16* k_h    = (_Float16*)alloc(BNC * 2);      // [B,256,N]
    _Float16* v_h    = (_Float16*)alloc(BNC * 2);      // [B,8,N,32]
    _Float16* q_h    = (_Float16*)alloc(BNC * 2);      // [B,256,N]
    _Float16* attn_h = (_Float16*)alloc(BNC * 2);      // [B,N,C]
    float*    qn2    = (float*)   alloc(Bq * Cq * 4);
    float*    kn2    = (float*)   alloc(Bq * Cq * 4);
    _Float16* P_h    = (_Float16*)alloc((size_t)Bq * HEADS * HD * HD * 2);
    _Float16* wkv_h  = (_Float16*)alloc((size_t)2 * Cq * Cq * 2);
    _Float16* wq_h   = (_Float16*)alloc((size_t)9 * Cq * Cq * 2);
    _Float16* wp_h   = (_Float16*)alloc((size_t)Cq * Cq * 2);

    // 1. weight repack
    cvt_w_kernel<<<(Cq * Cq * 9 + 255) / 256, 256, 0, stream>>>(
        W_kv, W_q, W_proj, wkv_h, wq_h, wp_h);

    // 2. LayerNorms (x -> kv branch, y -> q branch + f32 residual copy)
    ln_kernel<<<Bq * Nq / 256, 256, 0, stream>>>(x, ln_kv_w, ln_kv_b, xn_h, nullptr);
    ln_kernel<<<Bq * Nq / 256, 256, 0, stream>>>(y, ln_q_w, ln_q_b, yn_h, yn_f);

    // 3. kv = W_kv @ xn  (M=512) -> f16 channel-major
    gemm1x1_kernel<<<(Bq * (512 / 16) * (Nq / 64) * 32) / 256, 256, 0, stream>>>(
        wkv_h, xn_h, nullptr, kv_h, nullptr, 2 * Cq);

    // 4. depthwise 3x3 -> k (ch-major) and v ([b,h,n,d])
    dw_kernel<<<(size_t)Bq * 2 * Cq * Nq / 256, 256, 0, stream>>>(kv_h, W_dw, k_h, v_h);

    // 5. q = conv3x3(yn)  -> f16 channel-major
    qconv_kernel<<<(Bq * (Cq / 16) * (Nq / 64) * 32) / 256, 256, 0, stream>>>(
        wq_h, yn_h, q_h);

    // 6. row norms of q and k
    rowsumsq_kernel<<<Bq * Cq, 256, 0, stream>>>(q_h, qn2, Nq);
    rowsumsq_kernel<<<Bq * Cq, 256, 0, stream>>>(k_h, kn2, Nq);

    // 7. attention scores + softmax -> P
    attn_kernel<<<Bq * HEADS, 256, 0, stream>>>(q_h, k_h, qn2, kn2, temp, P_h);

    // 8. out = P @ V -> channels-last f16
    av_kernel<<<(Bq * HEADS * (Nq / 64) * 32) / 256, 256, 0, stream>>>(P_h, v_h, attn_h);

    // 9. final = yn + W_proj @ out  (f32, NCHW)
    gemm1x1_kernel<<<(Bq * (Cq / 16) * (Nq / 64) * 32) / 256, 256, 0, stream>>>(
        wp_h, attn_h, out, nullptr, yn_f, Cq);
}
